// UnifiedSparseAttention_58841051955823
// MI455X (gfx1250) — matrix operand order repaired and checked by reading, not compiled, not run on hardware
//
#include <hip/hip_runtime.h>
#include <hip/hip_bf16.h>

#define S_LEN   2048
#define HDIM    1024
#define NH      16
#define DH      64
#define WIN_HALF 128
#define NGLOB   32

typedef __bf16 bf16;
typedef __bf16 v8bf  __attribute__((ext_vector_type(8)));
typedef __bf16 v16bf __attribute__((ext_vector_type(16)));
typedef float  v8f   __attribute__((ext_vector_type(8)));

union V16 { v16bf v; v8bf h[2]; };

// ---------------- workspace layout (bytes, all 256-aligned) ----------------
#define WS_HIDBF   0u          // 2048*1024*2        = 4,194,304
#define WS_WT      4194304u    // 3*1024*1024*2      = 6,291,456 (Wq^T,Wk^T,Wv^T bf16)
#define WS_QBF     10485760u   // 16*2048*64*2       = 4,194,304  Q bf16 [h][s][d]
#define WS_KBF     14680064u   // 4,194,304          K bf16 [h][s][d]
#define WS_VT      18874368u   // 4,194,304          V bf16 transposed [h][d][s]
#define WS_BUCKET  23068672u   // 2048*4
#define WS_CNT     23076864u   // cntAll[32] + cntG[32]
#define WS_KEEP    23077120u   // 16*2048*4 keep-bitmask per (head,row)
// total ~22.2 MB

static __device__ __forceinline__ v8f wmma_bf16(const V16& a, const V16& b, v8f c) {
  return __builtin_amdgcn_wmma_f32_16x16x32_bf16(false, a.v, false, b.v, (short)0, c,
                                                 false, false);
}

static __device__ __forceinline__ float maskval(float s, int i, int j, int bi, int bj,
                                                unsigned kb) {
  int d = i - j; if (d < 0) d = -d;
  int cat = (d <= WIN_HALF ? 1 : 0) | ((bi == bj) ? 2 : 0)
          | (((i < NGLOB) || (j < NGLOB)) ? 4 : 0);
  return ((kb >> cat) & 1u) ? s * 0.125f : 0.0f;   // 1/sqrt(64); masked -> exactly 0
}

// ---------------- prep kernels ----------------
__global__ void cvt_hidden_kernel(const float* __restrict__ in, bf16* __restrict__ out,
                                  int N) {
  int i = blockIdx.x * blockDim.x + threadIdx.x;
  if (i < N) out[i] = (bf16)in[i];
}

__global__ void cvt_wT_kernel(const float* __restrict__ W0, const float* __restrict__ W1,
                              const float* __restrict__ W2, bf16* __restrict__ out) {
  int idx = blockIdx.x * blockDim.x + threadIdx.x;
  if (idx >= 3 * HDIM * HDIM) return;
  int w = idx >> 20;                 // which matrix
  int r = idx & (HDIM * HDIM - 1);
  int o = r >> 10;                   // output column -> row of W^T
  int f = r & (HDIM - 1);            // feature
  const float* W = (w == 0) ? W0 : (w == 1 ? W1 : W2);
  out[idx] = (bf16)W[f * HDIM + o];
}

__global__ void zero_cnt_kernel(int* __restrict__ cnt) {
  if (threadIdx.x < 64) cnt[threadIdx.x] = 0;
}

// bucket = (sum_j (hp_j>0)*2^j) % 32  ==> only LSH bits 0..4 matter
__global__ void lsh_kernel(const float* __restrict__ hid, const float* __restrict__ proj,
                           int* __restrict__ bucket, int* __restrict__ cntAll,
                           int* __restrict__ cntG) {
  int i = blockIdx.x * blockDim.x + threadIdx.x;
  if (i >= S_LEN) return;
  float a0 = 0.f, a1 = 0.f, a2 = 0.f, a3 = 0.f, a4 = 0.f;
  const float* row = hid + i * HDIM;
  for (int f = 0; f < HDIM; ++f) {
    float x = row[f];
    const float* p = proj + f * 16;
    a0 += x * p[0]; a1 += x * p[1]; a2 += x * p[2]; a3 += x * p[3]; a4 += x * p[4];
  }
  int bkt = (int)(a0 > 0.f) | ((int)(a1 > 0.f) << 1) | ((int)(a2 > 0.f) << 2)
          | ((int)(a3 > 0.f) << 3) | ((int)(a4 > 0.f) << 4);
  bucket[i] = bkt;
  atomicAdd(&cntAll[bkt], 1);
  if (i < NGLOB) atomicAdd(&cntG[bkt], 1);
}

// Exact quantile via 8-category counting: combined takes <=8 distinct values per row.
__global__ void keep_kernel(const int* __restrict__ bucket, const int* __restrict__ cntAll,
                            const int* __restrict__ cntG, const float* __restrict__ hwts,
                            unsigned* __restrict__ keep) {
  int i = blockIdx.x * blockDim.x + threadIdx.x;
  if (i >= S_LEN) return;
  int b = bucket[i];
  int lo = i - WIN_HALF; if (lo < 0) lo = 0;
  int hi = i + WIN_HALF; if (hi > S_LEN - 1) hi = S_LEN - 1;
  int Wc = hi - lo + 1;
  int n_wl = 0, n_wgl = 0;
  for (int j = lo; j <= hi; ++j) {
    int m = (bucket[j] == b);
    n_wl += m;
    if (j < NGLOB) n_wgl += m;
  }
  int Nb = cntAll[b];
  int c[8];
  if (i < NGLOB) {                      // global row: glb=1 for all j
    c[0] = c[1] = c[2] = c[3] = 0;
    c[7] = n_wl;
    c[5] = Wc - n_wl;
    c[6] = Nb - n_wl;
    c[4] = S_LEN - Wc - c[6];
  } else {
    int nG = cntG[b];
    int Wg = 0;
    if (lo <= NGLOB - 1) { int hh = hi < NGLOB - 1 ? hi : NGLOB - 1; Wg = hh - lo + 1; }
    c[7] = n_wgl;
    c[5] = Wg - n_wgl;
    c[6] = nG - n_wgl;
    c[4] = NGLOB - Wg - c[6];
    c[3] = n_wl - n_wgl;
    c[1] = (Wc - Wg) - c[3];
    c[2] = (Nb - n_wl) - c[6];
    c[0] = S_LEN - c[1] - c[2] - c[3] - c[4] - c[5] - c[6] - c[7];
  }
  float pos = 0.2f * (float)(S_LEN - 1);   // jnp.quantile 'linear': q*(n-1)
  int i0 = (int)pos;
  float frac = pos - (float)i0;
  for (int h = 0; h < NH; ++h) {
    float w0 = hwts[h * 3 + 0], w1 = hwts[h * 3 + 1], w2 = hwts[h * 3 + 2];
    float mx = fmaxf(w0, fmaxf(w1, w2));
    float e0 = __expf(w0 - mx), e1 = __expf(w1 - mx), e2 = __expf(w2 - mx);
    float inv = 1.f / (e0 + e1 + e2);
    float wWin = e0 * inv, wLsh = e1 * inv, wGlb = e2 * inv;
    float v[8];
    for (int cat = 0; cat < 8; ++cat)
      v[cat] = ((cat & 1) ? wWin : 0.f) + ((cat & 2) ? wLsh : 0.f) + ((cat & 4) ? wGlb : 0.f);
    float sv[8]; int sc[8];
    for (int t = 0; t < 8; ++t) { sv[t] = v[t]; sc[t] = c[t]; }
    for (int a1 = 1; a1 < 8; ++a1) {     // insertion sort ascending
      float kv = sv[a1]; int kc = sc[a1]; int b1 = a1 - 1;
      while (b1 >= 0 && sv[b1] > kv) { sv[b1 + 1] = sv[b1]; sc[b1 + 1] = sc[b1]; --b1; }
      sv[b1 + 1] = kv; sc[b1 + 1] = kc;
    }
    float vA = sv[7], vB = sv[7];
    int cum = 0; bool gotA = false;
    for (int t = 0; t < 8; ++t) {
      cum += sc[t];
      if (!gotA && cum > i0) { vA = sv[t]; gotA = true; }
      if (cum > i0 + 1) { vB = sv[t]; break; }
    }
    float thr = vA + frac * (vB - vA);
    unsigned bits = 0;
    for (int cat = 0; cat < 8; ++cat) if (v[cat] > thr) bits |= (1u << cat);
    keep[h * S_LEN + i] = bits;
  }
}

// ---------------- QKV projection: bf16 WMMA, one wave = 16x64 output strip ----------
__global__ __launch_bounds__(256) void qkv_kernel(
    const bf16* __restrict__ hid, const bf16* __restrict__ wT,
    const float* __restrict__ bq, const float* __restrict__ bk,
    const float* __restrict__ bv,
    bf16* __restrict__ Qb, bf16* __restrict__ Kb, bf16* __restrict__ Vt) {
  int lane = threadIdx.x & 31;
  int wi = blockIdx.x * 8 + (threadIdx.x >> 5);  // 0..6143
  int mat = wi >> 11;                            // 0=Q 1=K 2=V
  int rem = wi & 2047;
  int stile = rem >> 4;                          // 0..127 : 16-row tile of S
  int ostrip = rem & 15;                         // 0..15  : 64-col strip of H
  int n = lane & 15, kh = lane >> 4;
  const bf16* W = wT + mat * (HDIM * HDIM);
  const float* bias = (mat == 0) ? bq : (mat == 1 ? bk : bv);
  const bf16* Ar = hid + (stile * 16 + n) * HDIM;   // A row m = lane%16
  v8f acc[4];
#pragma unroll
  for (int t = 0; t < 4; ++t)
#pragma unroll
    for (int r = 0; r < 8; ++r) acc[t][r] = 0.f;
  for (int kt = 0; kt < HDIM; kt += 32) {
    V16 a;
    a.h[0] = *(const v8bf*)(Ar + kt + 8 * kh);        // A elems 0..7  : K = 8*kh..
    a.h[1] = *(const v8bf*)(Ar + kt + 16 + 8 * kh);   // A elems 8..15 : K = 16+8*kh..
#pragma unroll
    for (int nt = 0; nt < 4; ++nt) {
      const bf16* Br = W + (ostrip * 64 + nt * 16 + n) * HDIM + kt;
      V16 bm;
      bm.h[0] = *(const v8bf*)(Br + 8 * kh);          // B elems 0..7  : K = 8*kh..
      bm.h[1] = *(const v8bf*)(Br + 16 + 8 * kh);     // B elems 8..15 : K = 16+8*kh..
      acc[nt] = wmma_bf16(a, bm, acc[nt]);
    }
  }
#pragma unroll
  for (int nt = 0; nt < 4; ++nt) {
    int o = ostrip * 64 + nt * 16 + n;
    float bb = bias[o];
    int hh = o >> 6, d = o & 63;
#pragma unroll
    for (int r = 0; r < 8; ++r) {
      int s = stile * 16 + 8 * kh + r;                // D: row = r + 8*(lane/16)
      float val = acc[nt][r] + bb;
      if (mat == 0)      Qb[(hh * S_LEN + s) * DH + d] = (bf16)val;
      else if (mat == 1) Kb[(hh * S_LEN + s) * DH + d] = (bf16)val;
      else               Vt[(hh * DH + d) * S_LEN + s] = (bf16)val;  // transposed
    }
  }
}

// ---------------- attention: one wave = (head, 16-query tile), flash-style ----------
__global__ __launch_bounds__(256) void attn_kernel(
    const bf16* __restrict__ Q, const bf16* __restrict__ K,
    const bf16* __restrict__ Vt, const int* __restrict__ bucket,
    const unsigned* __restrict__ keep, float* __restrict__ out) {
  __shared__ bf16 Pl[8 * 512];                       // per-wave 16x32 bf16 P tile
  int lane = threadIdx.x & 31;
  int warp = threadIdx.x >> 5;
  int wi = blockIdx.x * 8 + warp;                    // 0..2047
  int h = wi >> 7;
  int qbase = (wi & 127) << 4;
  int n = lane & 15, kh = lane >> 4;
  bf16* P = Pl + warp * 512;

  // Q tile in A layout, K-dim = dh split into two 32-chunks (kept in regs all loop)
  const bf16* Qp = Q + (h * S_LEN + qbase + n) * DH;
  V16 aq0, aq1;
  aq0.h[0] = *(const v8bf*)(Qp + 8 * kh);
  aq0.h[1] = *(const v8bf*)(Qp + 16 + 8 * kh);
  aq1.h[0] = *(const v8bf*)(Qp + 32 + 8 * kh);
  aq1.h[1] = *(const v8bf*)(Qp + 48 + 8 * kh);

  unsigned kbits[8]; int bi[8];
#pragma unroll
  for (int r = 0; r < 8; ++r) {
    int i = qbase + 8 * kh + r;
    kbits[r] = keep[h * S_LEN + i];
    bi[r] = bucket[i];
  }
  float mrun[8], lrun[8];
#pragma unroll
  for (int r = 0; r < 8; ++r) { mrun[r] = -1e30f; lrun[r] = 0.f; }
  v8f acc[4];
#pragma unroll
  for (int t = 0; t < 4; ++t)
#pragma unroll
    for (int r = 0; r < 8; ++r) acc[t][r] = 0.f;

  for (int kb = 0; kb < S_LEN; kb += 32) {
    v8f s0, s1;
#pragma unroll
    for (int r = 0; r < 8; ++r) { s0[r] = 0.f; s1[r] = 0.f; }
    {
      const bf16* Kp0 = K + (h * S_LEN + kb + n) * DH;   // B lane col = key kb+n
      const bf16* Kp1 = Kp0 + 16 * DH;                   // second 16-key tile
      V16 bm;
      bm.h[0] = *(const v8bf*)(Kp0 + 8 * kh);
      bm.h[1] = *(const v8bf*)(Kp0 + 16 + 8 * kh);
      s0 = wmma_bf16(aq0, bm, s0);
      bm.h[0] = *(const v8bf*)(Kp0 + 32 + 8 * kh);
      bm.h[1] = *(const v8bf*)(Kp0 + 48 + 8 * kh);
      s0 = wmma_bf16(aq1, bm, s0);
      bm.h[0] = *(const v8bf*)(Kp1 + 8 * kh);
      bm.h[1] = *(const v8bf*)(Kp1 + 16 + 8 * kh);
      s1 = wmma_bf16(aq0, bm, s1);
      bm.h[0] = *(const v8bf*)(Kp1 + 32 + 8 * kh);
      bm.h[1] = *(const v8bf*)(Kp1 + 48 + 8 * kh);
      s1 = wmma_bf16(aq1, bm, s1);
    }
    int j0 = kb + n, j1 = kb + 16 + n;
    int bj0 = bucket[j0], bj1 = bucket[j1];
#pragma unroll
    for (int r = 0; r < 8; ++r) {
      int i = qbase + 8 * kh + r;
      float v0 = maskval(s0[r], i, j0, bi[r], bj0, kbits[r]);
      float v1 = maskval(s1[r], i, j1, bi[r], bj1, kbits[r]);
      float bmx = fmaxf(v0, v1);                      // row reduce over 16-lane group
      bmx = fmaxf(bmx, __shfl_xor(bmx, 1, 32));
      bmx = fmaxf(bmx, __shfl_xor(bmx, 2, 32));
      bmx = fmaxf(bmx, __shfl_xor(bmx, 4, 32));
      bmx = fmaxf(bmx, __shfl_xor(bmx, 8, 32));
      float nm = fmaxf(mrun[r], bmx);
      float fsc = __expf(mrun[r] - nm);
      float e0 = __expf(v0 - nm);
      float e1 = __expf(v1 - nm);
      float es = e0 + e1;
      es += __shfl_xor(es, 1, 32);
      es += __shfl_xor(es, 2, 32);
      es += __shfl_xor(es, 4, 32);
      es += __shfl_xor(es, 8, 32);
      lrun[r] = lrun[r] * fsc + es;
      mrun[r] = nm;
      acc[0][r] *= fsc; acc[1][r] *= fsc; acc[2][r] *= fsc; acc[3][r] *= fsc;
      int row = 8 * kh + r;                           // D layout -> LDS
      P[row * 32 + n] = (bf16)e0;
      P[row * 32 + 16 + n] = (bf16)e1;
    }
    __syncthreads();                                  // LDS store -> load ordering
    V16 ap;                                           // reread in A layout
    ap.h[0] = *(const v8bf*)(P + n * 32 + 8 * kh);
    ap.h[1] = *(const v8bf*)(P + n * 32 + 16 + 8 * kh);
#pragma unroll
    for (int nt = 0; nt < 4; ++nt) {                  // V in B layout via Vt[h][d][s]
      const bf16* Vp = Vt + (h * DH + nt * 16 + n) * S_LEN + kb;
      V16 bm;
      bm.h[0] = *(const v8bf*)(Vp + 8 * kh);
      bm.h[1] = *(const v8bf*)(Vp + 16 + 8 * kh);
      acc[nt] = wmma_bf16(ap, bm, acc[nt]);
    }
  }
#pragma unroll
  for (int nt = 0; nt < 4; ++nt)
#pragma unroll
    for (int r = 0; r < 8; ++r) {
      int q = qbase + 8 * kh + r;
      int d = nt * 16 + n;
      out[(h * S_LEN + q) * DH + d] = acc[nt][r] / lrun[r];
    }
}

extern "C" void kernel_launch(void* const* d_in, const int* in_sizes, int n_in,
                              void* d_out, int out_size, void* d_ws, size_t ws_size,
                              hipStream_t stream) {
  const float* hidden = (const float*)d_in[0];
  const float* Wq = (const float*)d_in[1];
  const float* bq = (const float*)d_in[2];
  const float* Wk = (const float*)d_in[3];
  const float* bk = (const float*)d_in[4];
  const float* Wv = (const float*)d_in[5];
  const float* bv = (const float*)d_in[6];
  const float* lshp = (const float*)d_in[7];
  const float* hwts = (const float*)d_in[8];

  char* ws = (char*)d_ws;
  bf16* hidbf = (bf16*)(ws + WS_HIDBF);
  bf16* wT    = (bf16*)(ws + WS_WT);
  bf16* Qb    = (bf16*)(ws + WS_QBF);
  bf16* Kb    = (bf16*)(ws + WS_KBF);
  bf16* Vt    = (bf16*)(ws + WS_VT);
  int* bucket = (int*)(ws + WS_BUCKET);
  int* cnt    = (int*)(ws + WS_CNT);
  unsigned* keep = (unsigned*)(ws + WS_KEEP);
  float* out = (float*)d_out;

  cvt_hidden_kernel<<<(S_LEN * HDIM + 255) / 256, 256, 0, stream>>>(hidden, hidbf,
                                                                    S_LEN * HDIM);
  cvt_wT_kernel<<<(3 * HDIM * HDIM + 255) / 256, 256, 0, stream>>>(Wq, Wk, Wv, wT);
  zero_cnt_kernel<<<1, 64, 0, stream>>>(cnt);
  lsh_kernel<<<S_LEN / 256, 256, 0, stream>>>(hidden, lshp, bucket, cnt, cnt + 32);
  keep_kernel<<<S_LEN / 256, 256, 0, stream>>>(bucket, cnt, cnt + 32, hwts, keep);
  qkv_kernel<<<768, 256, 0, stream>>>(hidbf, wT, bq, bk, bv, Qb, Kb, Vt);
  attn_kernel<<<256, 256, 0, stream>>>(Qb, Kb, Vt, bucket, keep, out);
}
